// MSA_90125593739844
// MI455X (gfx1250) — matrix operand order repaired and checked by reading, not compiled, not run on hardware
//
#include <hip/hip_runtime.h>

// ---------------------------------------------------------------------------
// Problem constants (match the JAX reference)
// ---------------------------------------------------------------------------
#define BB   64
#define AA   26
#define LL   384
#define DD   512
#define KK   18
#define CK   (AA*KK)          // 468 contraction length for conv-as-GEMM
#define CKP  480              // padded to multiple of 32
#define GAPP 0.0f
#define TT   1.0f

typedef _Float16 v16h __attribute__((ext_vector_type(16)));
typedef _Float16 v8h  __attribute__((ext_vector_type(8)));
typedef float    v8f  __attribute__((ext_vector_type(8)));

#define WMMA_F16(A, B, C) \
  __builtin_amdgcn_wmma_f32_16x16x32_f16(false, (A), false, (B), (short)0, (C), false, false)

// ---------------------------------------------------------------------------
// Kernel 1: Conv1d('same', even kernel K=18) as WMMA GEMM.
//   emb[b,l,d] = bias[d] + sum_c x[b, c/18, l + c%18 - 8] * w[d*468 + c]
// One wave computes a 16(L) x 64(D) tile: 4 f32 accumulators share the A
// (im2col) operand; B is 16 contiguous floats/lane -> float4 vector loads.
// ---------------------------------------------------------------------------
__global__ void conv_wmma(const float* __restrict__ x,    // (B, A, L)
                          const float* __restrict__ w,    // (D, A, K) = (D, 468)
                          const float* __restrict__ bias, // (D,)
                          _Float16* __restrict__ emb)     // (B, L, D) f16
{
  const int wave = (blockIdx.x * blockDim.x + threadIdx.x) >> 5;
  const int lane = threadIdx.x & 31;
  const int tilesD = DD / 64, tilesL = LL / 16;   // 8, 24
  int t = wave;
  const int dt = t % tilesD; t /= tilesD;
  const int lt = t % tilesL; t /= tilesL;
  const int b  = t;
  if (b >= BB) return;
  const int l0 = lt * 16, d0 = dt * 64;

  const int m   = lane & 15;   // A row (output L position in tile)
  const int khi = lane >> 4;   // A K-half select (contiguous k = 16*khi + hv)
  const int n   = lane & 15;   // B col (output D position in sub-tile)
  const int kg  = lane >> 4;   // B K-group select

  const float* __restrict__ xb  = x + (size_t)b * AA * LL;
  const float* __restrict__ wr0 = w + (size_t)(d0 +  0 + n) * CK;
  const float* __restrict__ wr1 = w + (size_t)(d0 + 16 + n) * CK;
  const float* __restrict__ wr2 = w + (size_t)(d0 + 32 + n) * CK;
  const float* __restrict__ wr3 = w + (size_t)(d0 + 48 + n) * CK;

  v8f acc0 = {}, acc1 = {}, acc2 = {}, acc3 = {};
  for (int c0 = 0; c0 < CKP; c0 += 32) {
    // ---- A operand: im2col gather (c -> (a,k) decode only here)
    v16h av;
#pragma unroll
    for (int hv = 0; hv < 16; ++hv) {
      const int v = hv >> 1, h = hv & 1;
      const int kla = (khi << 4) + hv;
      const int ca = c0 + kla;
      float va = 0.f;
      if (ca < CK) {
        const int a = ca / KK, k = ca - a * KK;
        const int pos = l0 + m + k - ((KK - 1) / 2);
        if (pos >= 0 && pos < LL) va = xb[(size_t)a * LL + pos];
      }
      av[hv] = (_Float16)va;
    }
    // ---- B operand: 16 contiguous floats per lane (row offset == c)
    const int base = c0 + (kg << 4);
    v16h bv0, bv1, bv2, bv3;
    if (base + 16 <= CK) {
#pragma unroll
      for (int q = 0; q < 4; ++q) {
        const float4 f0 = ((const float4*)(wr0 + base))[q];
        const float4 f1 = ((const float4*)(wr1 + base))[q];
        const float4 f2 = ((const float4*)(wr2 + base))[q];
        const float4 f3 = ((const float4*)(wr3 + base))[q];
        bv0[4*q+0]=(_Float16)f0.x; bv0[4*q+1]=(_Float16)f0.y; bv0[4*q+2]=(_Float16)f0.z; bv0[4*q+3]=(_Float16)f0.w;
        bv1[4*q+0]=(_Float16)f1.x; bv1[4*q+1]=(_Float16)f1.y; bv1[4*q+2]=(_Float16)f1.z; bv1[4*q+3]=(_Float16)f1.w;
        bv2[4*q+0]=(_Float16)f2.x; bv2[4*q+1]=(_Float16)f2.y; bv2[4*q+2]=(_Float16)f2.z; bv2[4*q+3]=(_Float16)f2.w;
        bv3[4*q+0]=(_Float16)f3.x; bv3[4*q+1]=(_Float16)f3.y; bv3[4*q+2]=(_Float16)f3.z; bv3[4*q+3]=(_Float16)f3.w;
      }
    } else {  // tail chunk (c0 = 448): guard 468..479
#pragma unroll
      for (int hv = 0; hv < 16; ++hv) {
        const int cb = base + hv;
        const bool ok = cb < CK;
        bv0[hv] = (_Float16)(ok ? wr0[cb] : 0.f);
        bv1[hv] = (_Float16)(ok ? wr1[cb] : 0.f);
        bv2[hv] = (_Float16)(ok ? wr2[cb] : 0.f);
        bv3[hv] = (_Float16)(ok ? wr3[cb] : 0.f);
      }
    }
    acc0 = WMMA_F16(av, bv0, acc0);
    acc1 = WMMA_F16(av, bv1, acc1);
    acc2 = WMMA_F16(av, bv2, acc2);
    acc3 = WMMA_F16(av, bv3, acc3);
  }
  // C/D layout: VGPR r -> M = r + 8*(lane>=16); N = lane&15
  const int ncol = lane & 15;
  const int mhi  = (lane >> 4) * 8;
  const float bs0 = bias[d0 +  0 + ncol];
  const float bs1 = bias[d0 + 16 + ncol];
  const float bs2 = bias[d0 + 32 + ncol];
  const float bs3 = bias[d0 + 48 + ncol];
#pragma unroll
  for (int r = 0; r < 8; ++r) {
    _Float16* row = emb + ((size_t)b * LL + (l0 + r + mhi)) * DD + ncol;
    row[d0 +  0] = (_Float16)(acc0[r] + bs0);
    row[d0 + 16] = (_Float16)(acc1[r] + bs1);
    row[d0 + 32] = (_Float16)(acc2[r] + bs2);
    row[d0 + 48] = (_Float16)(acc3[r] + bs3);
  }
}

// ---------------------------------------------------------------------------
// Kernel 2: sim[b,i,j] = sum_d e[b,i,d] * e[0,j,d]  (WMMA f16-in/f32-acc)
// One wave computes a 16(i) x 64(j) tile: 4 accumulators share the A operand.
// A/B lane layouts give contiguous 8-half runs -> 16-byte vector loads.
// ---------------------------------------------------------------------------
__global__ void sim_wmma(const _Float16* __restrict__ emb,  // (B, L, D) f16
                         float* __restrict__ sim)           // (B, L, L)
{
  const int wave = (blockIdx.x * blockDim.x + threadIdx.x) >> 5;
  const int lane = threadIdx.x & 31;
  const int tJ = LL / 64, tI = LL / 16;   // 6, 24
  int t = wave;
  const int jt = t % tJ; t /= tJ;
  const int it = t % tI; t /= tI;
  const int b  = t;
  if (b >= BB) return;
  const int i0 = it * 16, j0 = jt * 64;

  const int m   = lane & 15;
  const int khi = lane >> 4;
  const int n   = lane & 15;
  const int kg  = lane >> 4;

  const _Float16* __restrict__ arow = emb + ((size_t)b * LL + (i0 + m)) * DD;
  const _Float16* __restrict__ br0  = emb + (size_t)(j0 +  0 + n) * DD;  // batch 0
  const _Float16* __restrict__ br1  = emb + (size_t)(j0 + 16 + n) * DD;
  const _Float16* __restrict__ br2  = emb + (size_t)(j0 + 32 + n) * DD;
  const _Float16* __restrict__ br3  = emb + (size_t)(j0 + 48 + n) * DD;

  v8f acc0 = {}, acc1 = {}, acc2 = {}, acc3 = {};
  for (int d0 = 0; d0 < DD; d0 += 32) {
    union U { v16h v; v8h h[2]; };
    U A, B0, B1, B2, B3;
    A.h[0]  = *(const v8h*)(arow + d0 + 16 * khi);      // k = 16*khi + [0..7]
    A.h[1]  = *(const v8h*)(arow + d0 + 16 * khi + 8);  // k = 16*khi+8 + [0..7]
    B0.h[0] = *(const v8h*)(br0 + d0 + 16 * kg);
    B0.h[1] = *(const v8h*)(br0 + d0 + 16 * kg + 8);
    B1.h[0] = *(const v8h*)(br1 + d0 + 16 * kg);
    B1.h[1] = *(const v8h*)(br1 + d0 + 16 * kg + 8);
    B2.h[0] = *(const v8h*)(br2 + d0 + 16 * kg);
    B2.h[1] = *(const v8h*)(br2 + d0 + 16 * kg + 8);
    B3.h[0] = *(const v8h*)(br3 + d0 + 16 * kg);
    B3.h[1] = *(const v8h*)(br3 + d0 + 16 * kg + 8);
    if (d0 + 32 < DD) {
      __builtin_prefetch(arow + d0 + 32, 0, 1);   // -> global_prefetch_b8
      __builtin_prefetch(br0 + d0 + 32, 0, 1);
      __builtin_prefetch(br2 + d0 + 32, 0, 1);
    }
    acc0 = WMMA_F16(A.v, B0.v, acc0);
    acc1 = WMMA_F16(A.v, B1.v, acc1);
    acc2 = WMMA_F16(A.v, B2.v, acc2);
    acc3 = WMMA_F16(A.v, B3.v, acc3);
  }
  const int ncol = lane & 15;
  const int mhi  = (lane >> 4) * 8;
#pragma unroll
  for (int r = 0; r < 8; ++r) {
    float* row = sim + ((size_t)b * LL + (i0 + r + mhi)) * LL + ncol;
    row[j0 +  0] = acc0[r];
    row[j0 + 16] = acc1[r];
    row[j0 + 32] = acc2[r];
    row[j0 + 48] = acc3[r];
  }
}

// ---------------------------------------------------------------------------
// Kernel 3: smoothed NW forward DP. LDS rotating anti-diagonal buffers
// (3 x 385 floats); one workgroup (384 thr = 12 wave32) per batch; one
// s_barrier per step. Interior V stored compacted (L x L) to global for bwd
// (fire-and-forget; borders are the closed form -g*idx, recomputed inline).
// ---------------------------------------------------------------------------
__global__ void __launch_bounds__(LL) nw_fwd(const float* __restrict__ sim,
                                             float* __restrict__ Vm)  // (B,L,L)
{
  __shared__ float dbuf[3][LL + 1];   // dbuf[s%3][i] = V[i][s-i], interior only
  const int b   = blockIdx.x;
  const int tid = threadIdx.x;
  const float* __restrict__ sb = sim + (size_t)b * LL * LL;
  float* __restrict__ Vb = Vm + (size_t)b * LL * LL;

  for (int s = 2; s <= 2 * LL; ++s) {
    const int ilo = max(1, s - LL), ihi = min(LL, s - 1);
    const int i = ilo + tid;
    if (i <= ihi) {
      const int j = s - i;
      const float vd = (i == 1) ? (-GAPP * (float)(j - 1))
                     : (j == 1) ? (-GAPP * (float)(i - 1))
                                : dbuf[(s - 2) % 3][i - 1];
      const float vu = ((i == 1) ? (-GAPP * (float)j) : dbuf[(s - 1) % 3][i - 1]) - GAPP;
      const float vl = ((j == 1) ? (-GAPP * (float)i) : dbuf[(s - 1) % 3][i]) - GAPP;
      const float mx = fmaxf(vd, fmaxf(vu, vl));
      const float sm = mx + TT * __logf(__expf((vd - mx) / TT) +
                                        __expf((vu - mx) / TT) +
                                        __expf((vl - mx) / TT));
      const float vij = sb[(size_t)(i - 1) * LL + (j - 1)] + sm;
      dbuf[s % 3][i] = vij;                         // recurrence state: LDS
      Vb[(size_t)(i - 1) * LL + (j - 1)] = vij;     // for backward: global
    }
    __syncthreads();
  }
}

// ---------------------------------------------------------------------------
// Kernel 4: backward wavefront. E[i,j] = dScore/dV[i,j] = align[i-1,j-1].
// E/V/sim anti-diagonals in rotating LDS (3-deep each, ~14 KB); each global
// V/sim cell is loaded exactly once. One s_barrier per step.
// ---------------------------------------------------------------------------
__global__ void __launch_bounds__(LL) nw_bwd(const float* __restrict__ sim,
                                             const float* __restrict__ Vm,
                                             const int* __restrict__ shapes,
                                             float* __restrict__ E)  // (B,L,L)
{
  __shared__ float Ebuf[3][LL + 1];
  __shared__ float Vbuf[3][LL + 1];
  __shared__ float Sbuf[3][LL + 1];
  const int b   = blockIdx.x;
  const int tid = threadIdx.x;
  const float* __restrict__ sb = sim + (size_t)b * LL * LL;
  const float* __restrict__ Vb = Vm + (size_t)b * LL * LL;
  float* __restrict__ Eb = E + (size_t)b * LL * LL;
  const int li = min(max(shapes[2 * b + 0], 0), LL);
  const int lj = min(max(shapes[2 * b + 1], 0), LL);

  for (int s = 2 * LL; s >= 2; --s) {
    const int ilo = max(1, s - LL), ihi = min(LL, s - 1);
    const int i = ilo + tid;
    if (i <= ihi) {
      const int j = s - i;
      const size_t o = (size_t)(i - 1) * LL + (j - 1);
      const float vij = Vb[o];       // loaded once; cached in LDS for parents
      const float sij = sb[o];
      Vbuf[s % 3][i] = vij;
      Sbuf[s % 3][i] = sij;
      float acc = (i == li && j == lj) ? 1.0f : 0.0f;
      if (i < LL && j < LL) {        // diagonal child (i+1, j+1) on diag s+2
        const int sl = (s + 2) % 3;
        const float smq = Vbuf[sl][i + 1] - Sbuf[sl][i + 1];
        acc += Ebuf[sl][i + 1] * __expf((vij - smq) / TT);
      }
      if (i < LL) {                  // "up" child (i+1, j) on diag s+1
        const int sl = (s + 1) % 3;
        const float smq = Vbuf[sl][i + 1] - Sbuf[sl][i + 1];
        acc += Ebuf[sl][i + 1] * __expf((vij - GAPP - smq) / TT);
      }
      if (j < LL) {                  // "left" child (i, j+1) on diag s+1
        const int sl = (s + 1) % 3;
        const float smq = Vbuf[sl][i] - Sbuf[sl][i];
        acc += Ebuf[sl][i] * __expf((vij - GAPP - smq) / TT);
      }
      Ebuf[s % 3][i] = acc;          // recurrence state: LDS
      Eb[o] = acc;                   // align output: global (read later)
    }
    __syncthreads();
  }
}

// ---------------------------------------------------------------------------
// Kernel 5: consensus[j,a] = (1/B) * sum_n sum_i matrices[n,a,i] * align[n,i,j]
// Lane mapping: consecutive lanes take consecutive j -> coalesced align reads.
// ---------------------------------------------------------------------------
__global__ void consensus_k(const float* __restrict__ x,   // (B, A, L)
                            const float* __restrict__ E,   // (B, L, L)
                            float* __restrict__ cons)      // (L, A)
{
  const int idx = blockIdx.x * blockDim.x + threadIdx.x;
  if (idx >= LL * AA) return;
  const int j = idx % LL, a = idx / LL;
  float acc = 0.f;
  for (int nb = 0; nb < BB; ++nb) {
    const float* __restrict__ xr = x + ((size_t)nb * AA + a) * LL;
    const float* __restrict__ Er = E + (size_t)nb * LL * LL + j;
    for (int i = 0; i < LL; ++i)
      acc = fmaf(xr[i], Er[(size_t)i * LL], acc);
  }
  cons[(size_t)j * AA + a] = acc * (1.0f / (float)BB);
}

// ---------------------------------------------------------------------------
// Kernel 6: out[n,i,a] = sum_j consensus[j,a] * align[n,i,j]
// ---------------------------------------------------------------------------
__global__ void out_k(const float* __restrict__ cons,   // (L, A)
                      const float* __restrict__ E,      // (B, L, L)
                      float* __restrict__ out)          // (B, L, A)
{
  const int idx = blockIdx.x * blockDim.x + threadIdx.x;
  if (idx >= BB * LL * AA) return;
  const int a  = idx % AA;
  const int t  = idx / AA;
  const int i  = t % LL;
  const int nb = t / LL;
  const float* __restrict__ Er = E + ((size_t)nb * LL + i) * LL;
  float acc = 0.f;
  for (int j = 0; j < LL; ++j)
    acc = fmaf(cons[(size_t)j * AA + a], Er[j], acc);
  out[idx] = acc;
}

// ---------------------------------------------------------------------------
// Launch: inputs {matrices, shapes, conv_w, conv_b}; output (B, L, A) f32.
// Workspace (~138 MB, L2-resident on MI455X's 192 MB L2):
//   emb f16 | sim f32 | V f32 (interior, LxL) | align f32 | consensus f32
// ---------------------------------------------------------------------------
extern "C" void kernel_launch(void* const* d_in, const int* in_sizes, int n_in,
                              void* d_out, int out_size, void* d_ws, size_t ws_size,
                              hipStream_t stream)
{
  (void)in_sizes; (void)n_in; (void)out_size; (void)ws_size;
  const float* x      = (const float*)d_in[0];
  const int*   shapes = (const int*)d_in[1];
  const float* w      = (const float*)d_in[2];
  const float* bias   = (const float*)d_in[3];
  float* out = (float*)d_out;

  char* ws = (char*)d_ws;
  size_t off = 0;
  auto alloc = [&](size_t bytes) {
    size_t p = off;
    off = (off + bytes + 255) & ~(size_t)255;
    return p;
  };
  _Float16* emb  = (_Float16*)(ws + alloc((size_t)BB * LL * DD * sizeof(_Float16)));
  float*    sim  = (float*)(ws + alloc((size_t)BB * LL * LL * sizeof(float)));
  float*    Vm   = (float*)(ws + alloc((size_t)BB * LL * LL * sizeof(float)));
  float*    E    = (float*)(ws + alloc((size_t)BB * LL * LL * sizeof(float)));
  float*    cons = (float*)(ws + alloc((size_t)LL * AA * sizeof(float)));

  {
    const int waves = BB * (LL / 16) * (DD / 64);       // 12288 tiles (16x64)
    const int thr = 256;                                // 8 wave32 per block
    conv_wmma<<<(waves * 32 + thr - 1) / thr, thr, 0, stream>>>(x, w, bias, emb);
  }
  {
    const int waves = BB * (LL / 16) * (LL / 64);       // 9216 tiles (16x64)
    const int thr = 256;
    sim_wmma<<<(waves * 32 + thr - 1) / thr, thr, 0, stream>>>(emb, sim);
  }
  nw_fwd<<<BB, LL, 0, stream>>>(sim, Vm);
  nw_bwd<<<BB, LL, 0, stream>>>(sim, Vm, shapes, E);
  {
    const int n = LL * AA;
    consensus_k<<<(n + 255) / 256, 256, 0, stream>>>(x, E, cons);
  }
  {
    const int n = BB * LL * AA;
    out_k<<<(n + 255) / 256, 256, 0, stream>>>(cons, E, out);
  }
}
